// CrossModalAttention_33097017983177
// MI455X (gfx1250) — compile-verified, matrix operand order checked by reading, not run on hardware
//
#include <hip/hip_runtime.h>
#include <hip/hip_bf16.h>

// ---------------------------------------------------------------------------
// CrossModalAttention fused WMMA implementation for gfx1250 (MI455X, wave32).
// B=8, C=256, H=W=64 -> N=4096, CR=32.
// Flash-attention style: 537MB score matrix never materialized.
// K tiles are DMA'd global->LDS with GLOBAL_LOAD_ASYNC_TO_LDS_B128 (ASYNCcnt),
// double-buffered, shared by all 8 waves of the block. V streams from L2.
// ---------------------------------------------------------------------------

#define BATCH 8
#define CH    256
#define CRD   32
#define NTOK  4096   // 64*64

typedef __attribute__((ext_vector_type(16))) _Float16 v16h;
typedef __attribute__((ext_vector_type(8)))  _Float16 v8h;
typedef __attribute__((ext_vector_type(8)))  float    v8f;

union V16U { v16h v; v8h h[2]; };

// D = A(16x32 f16) * B(32x16 f16) + C(16x16 f32)
__device__ __forceinline__ v8f wmma16x16x32(v16h a, v16h b, v8f c) {
    return __builtin_amdgcn_wmma_f32_16x16x32_f16(
        /*neg_a=*/false, a, /*neg_b=*/false, b,
        /*c_mod=*/(short)0, c, /*reuse_a=*/false, /*reuse_b=*/false);
}

// Load a 16x32 f16 tile in WMMA A-matrix layout (also valid as the B-matrix
// when the source holds B transposed, i.e. row index == N-column, k contiguous).
// ISA layout (CDNA5 ISA): lanes 0-15 hold row M=lane with K={0..7,16..23},
// lanes 16-31 hold row M=lane-16 with K={8..15,24..31}; 2 halves per VGPR.
__device__ __forceinline__ v16h load_tile16x32(const _Float16* base, int ld,
                                               int row0, int k0) {
    int lane = threadIdx.x & 31;
    int r    = row0 + (lane & 15);
    int kb   = k0 + ((lane & 16) ? 8 : 0);
    const _Float16* p = base + (size_t)r * ld + kb;
    V16U t;
    t.h[0] = *(const v8h*)(p);        // K = kb .. kb+7
    t.h[1] = *(const v8h*)(p + 16);   // K = kb+16 .. kb+23
    return t.v;
}

// Async DMA one 32x32 f16 K chunk (2KB, row-major) from global to LDS.
// One wave: each lane copies 64B as 4x b128. Tracked by ASYNCcnt.
__device__ __forceinline__ void async_copy_k(const _Float16* src, int j0,
                                             unsigned ldsoff, int lane) {
    unsigned long long ga =
        (unsigned long long)(uintptr_t)(src + (size_t)j0 * CRD) + (unsigned)lane * 64u;
    unsigned lo = ldsoff + (unsigned)lane * 64u;
    #pragma unroll
    for (int q = 0; q < 4; ++q) {
        asm volatile("global_load_async_to_lds_b128 %0, %1, off"
                     :: "v"(lo + (unsigned)q * 16u),
                        "v"(ga + (unsigned long long)q * 16ull)
                     : "memory");
    }
}

__device__ __forceinline__ void wait_asynccnt_le4() {
    asm volatile("s_wait_asynccnt 4" ::: "memory");
}

// ---------------------------------------------------------------------------
// 1) (B,C,N) f32 -> (B,N,C) f16 transposed convert through LDS tiles.
// ---------------------------------------------------------------------------
__global__ __launch_bounds__(256) void k_transpose_cvt(const float* __restrict__ in,
                                                       _Float16* __restrict__ outT) {
    __shared__ float tile[32][33];
    int b  = blockIdx.z;
    int c0 = blockIdx.y * 32;
    int n0 = blockIdx.x * 32;
    int tx = threadIdx.x & 31;
    int ty = threadIdx.x >> 5;                  // 0..7
    const float* inb = in + (size_t)b * CH * NTOK;
    _Float16*    ob  = outT + (size_t)b * NTOK * CH;
    #pragma unroll
    for (int yy = 0; yy < 32; yy += 8)
        tile[ty + yy][tx] = inb[(size_t)(c0 + ty + yy) * NTOK + (n0 + tx)];
    __syncthreads();
    #pragma unroll
    for (int yy = 0; yy < 32; yy += 8)
        ob[(size_t)(n0 + ty + yy) * CH + (c0 + tx)] = (_Float16)tile[tx][ty + yy];
}

// ---------------------------------------------------------------------------
// 2) flat f32 -> f16 (weights)
// ---------------------------------------------------------------------------
__global__ void k_cvt_f16(const float* __restrict__ in, _Float16* __restrict__ out, int n) {
    int i = blockIdx.x * blockDim.x + threadIdx.x;
    if (i < n) out[i] = (_Float16)in[i];
}

// ---------------------------------------------------------------------------
// 3) Projection GEMM: out[o,n] = sum_c W[o,c]*in[n,c] + bias[o]
//    W: (O,CH) f16 row-major.  in: (B,N,CH) f16.  One wave per 16x16 tile.
//    transposed_out=1 -> store (B,N,O) (for Q,K); 0 -> store (B,O,N) (for V).
// ---------------------------------------------------------------------------
__global__ __launch_bounds__(32) void k_proj(const _Float16* __restrict__ wh,
                                             const float* __restrict__ bias,
                                             const _Float16* __restrict__ inh,
                                             _Float16* __restrict__ outh,
                                             int O, int transposed_out) {
    int b  = blockIdx.z;
    int o0 = blockIdx.y * 16;
    int n0 = blockIdx.x * 16;
    const _Float16* inb = inh + (size_t)b * NTOK * CH;
    v8f acc = {};
    #pragma unroll
    for (int cc = 0; cc < CH; cc += 32) {
        v16h a  = load_tile16x32(wh,  CH, o0, cc);
        v16h bt = load_tile16x32(inb, CH, n0, cc);
        acc = wmma16x16x32(a, bt, acc);
    }
    int lane = threadIdx.x & 31;
    int hi   = (lane & 16) >> 4;
    int n    = n0 + (lane & 15);
    #pragma unroll
    for (int r = 0; r < 8; r++) {
        int o = o0 + r + (hi << 3);
        float v = acc[r] + bias[o];
        if (transposed_out)
            outh[((size_t)b * NTOK + n) * O + o] = (_Float16)v;
        else
            outh[((size_t)b * O + o) * NTOK + n] = (_Float16)v;
    }
}

// ---------------------------------------------------------------------------
// 4) Fused flash-attention.
//    Block = 1 batch x 16 query rows; 8 waves; wave w owns channels [32w,32w+32).
//    K chunks (32x32 f16) are async-DMA'd into double-buffered LDS by wave 0
//    and shared by all waves. Per j-chunk: 2 WMMA for S = Q K^T (CR=32 == one
//    WMMA K step), online softmax (16-lane shfl reductions), P staged through
//    per-wave LDS for D->A re-layout, then 2 WMMA for acc += P V.
// ---------------------------------------------------------------------------
__global__ __launch_bounds__(256) void k_attn(const _Float16* __restrict__ qT,
                                              const _Float16* __restrict__ kT,
                                              const _Float16* __restrict__ vh,
                                              _Float16* __restrict__ ah) {
    // Single shared block => LDS base offset 0 (needed for async dest offsets).
    // [0,2048)B   : K buffer 0 (32x32 f16)
    // [2048,4096)B: K buffer 1
    // [4096,12288)B: per-wave P staging (8 waves x 16x32 f16)
    __shared__ _Float16 smem[2 * 32 * 32 + 8 * 16 * 32];
    int b    = blockIdx.y;
    int i0   = blockIdx.x * 16;
    int wave = threadIdx.x >> 5;
    int lane = threadIdx.x & 31;
    int hi   = (lane & 16) >> 4;
    _Float16* kbuf0 = smem;
    _Float16* kbuf1 = smem + 1024;
    _Float16* pst   = smem + 2048 + wave * (16 * 32);

    const _Float16* qTb = qT + (size_t)b * NTOK * CRD;
    const _Float16* kTb = kT + (size_t)b * NTOK * CRD;
    const _Float16* vhb = vh + (size_t)b * CH * NTOK;

    // Q tile for this block's 16 rows: lives in registers the whole loop.
    v16h qa = load_tile16x32(qTb, CRD, i0, 0);

    int c0 = wave * 32;
    v8f acc0 = {};
    v8f acc1 = {};
    float m[8], s[8];
    #pragma unroll
    for (int r = 0; r < 8; r++) { m[r] = -1e30f; s[r] = 0.0f; }

    // Prime the K pipeline: two chunks in flight (4 async ops each).
    if (wave == 0) {
        async_copy_k(kTb, 0,  0u,    lane);
        async_copy_k(kTb, 32, 2048u, lane);
        wait_asynccnt_le4();          // chunk 0 resident (in-order completion)
    }
    __syncthreads();

    for (int t = 0; t < NTOK / 32; ++t) {
        int j0 = t * 32;
        const _Float16* kb = (t & 1) ? kbuf1 : kbuf0;

        v16h kb0 = load_tile16x32(kb, CRD, 0,  0);   // cols j0..j0+15
        v16h kb1 = load_tile16x32(kb, CRD, 16, 0);   // cols j0+16..j0+31
        v8f s0 = {};
        v8f s1 = {};
        s0 = wmma16x16x32(qa, kb0, s0);
        s1 = wmma16x16x32(qa, kb1, s1);

        // Online softmax. D layout: lane holds col j=lane&15 (tile0) / +16
        // (tile1); VGPR r holds row i = r + 8*(lane>=16). m/s replicated
        // within each 16-lane half.
        #pragma unroll
        for (int r = 0; r < 8; r++) {
            float v0 = s0[r], v1 = s1[r];
            float mx = fmaxf(v0, v1);
            mx = fmaxf(mx, __shfl_xor(mx, 1, 32));
            mx = fmaxf(mx, __shfl_xor(mx, 2, 32));
            mx = fmaxf(mx, __shfl_xor(mx, 4, 32));
            mx = fmaxf(mx, __shfl_xor(mx, 8, 32));
            float mn   = fmaxf(m[r], mx);
            float corr = __expf(m[r] - mn);
            float p0   = __expf(v0 - mn);
            float p1   = __expf(v1 - mn);
            float ps   = p0 + p1;
            ps += __shfl_xor(ps, 1, 32);
            ps += __shfl_xor(ps, 2, 32);
            ps += __shfl_xor(ps, 4, 32);
            ps += __shfl_xor(ps, 8, 32);
            s[r] = s[r] * corr + ps;
            m[r] = mn;
            acc0[r] *= corr;
            acc1[r] *= corr;
            int irow = r + (hi << 3);
            pst[irow * 32 + (lane & 15)]      = (_Float16)p0;
            pst[irow * 32 + 16 + (lane & 15)] = (_Float16)p1;
        }
        // Same-wave LDS RAW on pst: DS ops in-order; compiler inserts dscnt wait.
        v16h pa  = load_tile16x32(pst, 32, 0, 0);           // P in A layout
        v16h vb0 = load_tile16x32(vhb, NTOK, c0,      j0);  // V cols c0..c0+15
        v16h vb1 = load_tile16x32(vhb, NTOK, c0 + 16, j0);  // V cols c0+16..c0+31
        acc0 = wmma16x16x32(pa, vb0, acc0);
        acc1 = wmma16x16x32(pa, vb1, acc1);

        // Double-buffer handoff: everyone done reading kb, refill it for t+2,
        // then guarantee chunk t+1 is resident before the next iteration.
        __syncthreads();
        if (wave == 0) {
            if (t + 2 < NTOK / 32)
                async_copy_k(kTb, j0 + 64, (t & 1) ? 2048u : 0u, lane);
            wait_asynccnt_le4();
        }
        __syncthreads();
    }

    // Normalize and store attention output as (B, N, C) f16.
    _Float16* ahb = ah + (size_t)b * NTOK * CH;
    #pragma unroll
    for (int r = 0; r < 8; r++) {
        float inv = 1.0f / s[r];
        int i = i0 + r + (hi << 3);
        int c = c0 + (lane & 15);
        ahb[(size_t)i * CH + c]      = (_Float16)(acc0[r] * inv);
        ahb[(size_t)i * CH + c + 16] = (_Float16)(acc1[r] * inv);
    }
}

// ---------------------------------------------------------------------------
// 5) Output projection + residual: out[b,o,n] = gamma*(Wo@a + bo) + x[b,o,n]
// ---------------------------------------------------------------------------
__global__ __launch_bounds__(32) void k_oproj(const _Float16* __restrict__ woh,
                                              const float* __restrict__ bo,
                                              const _Float16* __restrict__ ah,
                                              const float* __restrict__ x,
                                              const float* __restrict__ gamma,
                                              float* __restrict__ out) {
    int b  = blockIdx.z;
    int o0 = blockIdx.y * 16;
    int n0 = blockIdx.x * 16;
    const _Float16* ab = ah + (size_t)b * NTOK * CH;
    v8f acc = {};
    #pragma unroll
    for (int cc = 0; cc < CH; cc += 32) {
        v16h a  = load_tile16x32(woh, CH, o0, cc);
        v16h bt = load_tile16x32(ab,  CH, n0, cc);
        acc = wmma16x16x32(a, bt, acc);
    }
    float g   = gamma[0];
    int lane  = threadIdx.x & 31;
    int hi    = (lane & 16) >> 4;
    int n     = n0 + (lane & 15);
    const float* xb = x   + (size_t)b * CH * NTOK;
    float*       ob = out + (size_t)b * CH * NTOK;
    #pragma unroll
    for (int r = 0; r < 8; r++) {
        int o = o0 + r + (hi << 3);
        size_t idx = (size_t)o * NTOK + n;
        ob[idx] = g * (acc[r] + bo[o]) + xb[idx];
    }
}

// ---------------------------------------------------------------------------
// Host-side launch
// ---------------------------------------------------------------------------
extern "C" void kernel_launch(void* const* d_in, const int* in_sizes, int n_in,
                              void* d_out, int out_size, void* d_ws, size_t ws_size,
                              hipStream_t stream) {
    const float* x     = (const float*)d_in[0];
    const float* ff    = (const float*)d_in[1];
    const float* wq    = (const float*)d_in[2];
    const float* bq    = (const float*)d_in[3];
    const float* wk    = (const float*)d_in[4];
    const float* bk    = (const float*)d_in[5];
    const float* wv    = (const float*)d_in[6];
    const float* bv    = (const float*)d_in[7];
    const float* wo    = (const float*)d_in[8];
    const float* bo    = (const float*)d_in[9];
    const float* gamma = (const float*)d_in[10];
    float* out = (float*)d_out;

    // Workspace partition (f16 buffers), 256B aligned.
    char* base = (char*)d_ws;
    size_t off = 0;
    auto alloc = [&](size_t elems) {
        _Float16* p = (_Float16*)(base + off);
        off += (elems * sizeof(_Float16) + 255) & ~(size_t)255;
        return p;
    };
    _Float16* xh  = alloc((size_t)BATCH * NTOK * CH);   // x  as (B,N,C)
    _Float16* ffh = alloc((size_t)BATCH * NTOK * CH);   // ff as (B,N,C)
    _Float16* wqh = alloc((size_t)CRD * CH);
    _Float16* wkh = alloc((size_t)CRD * CH);
    _Float16* wvh = alloc((size_t)CH * CH);
    _Float16* woh = alloc((size_t)CH * CH);
    _Float16* qT  = alloc((size_t)BATCH * NTOK * CRD);  // (B,N,CR)
    _Float16* kT  = alloc((size_t)BATCH * NTOK * CRD);  // (B,N,CR)
    _Float16* vh  = alloc((size_t)BATCH * CH * NTOK);   // (B,C,N)
    _Float16* ah  = alloc((size_t)BATCH * NTOK * CH);   // (B,N,C)

    // 1) transpose + convert inputs
    {
        dim3 g(NTOK / 32, CH / 32, BATCH);
        k_transpose_cvt<<<g, 256, 0, stream>>>(x,  xh);
        k_transpose_cvt<<<g, 256, 0, stream>>>(ff, ffh);
    }
    // 2) convert weights
    k_cvt_f16<<<(CRD * CH + 255) / 256, 256, 0, stream>>>(wq, wqh, CRD * CH);
    k_cvt_f16<<<(CRD * CH + 255) / 256, 256, 0, stream>>>(wk, wkh, CRD * CH);
    k_cvt_f16<<<(CH * CH + 255) / 256, 256, 0, stream>>>(wv, wvh, CH * CH);
    k_cvt_f16<<<(CH * CH + 255) / 256, 256, 0, stream>>>(wo, woh, CH * CH);

    // 3) projections
    k_proj<<<dim3(NTOK / 16, CRD / 16, BATCH), 32, 0, stream>>>(wqh, bq, xh,  qT, CRD, 1);
    k_proj<<<dim3(NTOK / 16, CRD / 16, BATCH), 32, 0, stream>>>(wkh, bk, ffh, kT, CRD, 1);
    k_proj<<<dim3(NTOK / 16, CH  / 16, BATCH), 32, 0, stream>>>(wvh, bv, ffh, vh, CH,  0);

    // 4) fused attention
    k_attn<<<dim3(NTOK / 16, BATCH), 256, 0, stream>>>(qT, kT, vh, ah);

    // 5) output projection + residual
    k_oproj<<<dim3(NTOK / 16, CH / 16, BATCH), 32, 0, stream>>>(woh, bo, ah, x, gamma, out);
}
